// AdvancedModel_10892037062902
// MI455X (gfx1250) — matrix operand order repaired and checked by reading, not compiled, not run on hardware
//
#include <hip/hip_runtime.h>
#include <hip/hip_bf16.h>

typedef __attribute__((ext_vector_type(16))) _Float16 v16h;
typedef __attribute__((ext_vector_type(8)))  float    v8f;
typedef __attribute__((ext_vector_type(4)))  unsigned int v4u;
typedef __attribute__((ext_vector_type(8)))  unsigned int v8u;

#define B_     8
#define C1     64
#define Hc     112
#define Wc     112
#define SP     (Hc*Wc)        // 12544
#define NPIX   (B_*SP)        // 100352
#define KDIM   576            // 9 taps * 64 channels, K order = tap*64 + c
#define NCHUNK 18             // 576 / 32
#define OC_DCN 128
#define TILE_N 64
#define LSTR   592            // 576 + 16 halves pad -> rows 32B aligned
#define AW_HALVES (8 * NCHUNK * 512)      // 73728 halves = 147456 B packed w_dcn
#define AW_QWORDS (AW_HALVES / 4)         // 18432 8-byte elements

// ---------------------------------------------------------------------------
// K0: pack w_off/w_msk (27 rows, zero-padded to 32) and w_dcn (128 rows) into
// a WMMA 16x16x32 f16 A-operand lane layout (contiguous K order, as B uses):
//   entry i = ((mtile*18 + chunk)*32 + lane)*16 + e
//   M = mtile*16 + (lane&15)
//   K = chunk*32 + (lane>>4)*16 + e,  tap = K>>6, c = K&63
// Also zeroes the pooled accumulator.
// ---------------------------------------------------------------------------
__global__ void pack_weights_k(const float* __restrict__ w_off,
                               const float* __restrict__ w_msk,
                               const float* __restrict__ w_dcn,
                               _Float16* __restrict__ Aom,
                               _Float16* __restrict__ Adcn,
                               float* __restrict__ pooled) {
  int i = blockIdx.x * blockDim.x + threadIdx.x;
  if (i < 8 * NCHUNK * 512) {
    int e = i & 15, lane = (i >> 4) & 31, chunk = (i >> 9) % NCHUNK, mtile = i / (NCHUNK * 512);
    int M = mtile * 16 + (lane & 15);
    int K = chunk * 32 + ((lane >> 4) << 4) + e;
    int tap = K >> 6, c = K & 63;
    Adcn[i] = (_Float16)w_dcn[(M * 64 + c) * 9 + tap];
  }
  if (i < 2 * NCHUNK * 512) {
    int e = i & 15, lane = (i >> 4) & 31, chunk = (i >> 9) % NCHUNK, mtile = i / (NCHUNK * 512);
    int M = mtile * 16 + (lane & 15);
    int K = chunk * 32 + ((lane >> 4) << 4) + e;
    int tap = K >> 6, c = K & 63;
    float w = 0.f;
    if (M < 18)      w = w_off[(M * 64 + c) * 9 + tap];
    else if (M < 27) w = w_msk[((M - 18) * 64 + c) * 9 + tap];
    Aom[i] = (_Float16)w;
  }
  if (i < B_ * OC_DCN) pooled[i] = 0.f;
}

// ---------------------------------------------------------------------------
// K1: 7x7/s2/p3 conv + BN + ReLU. One thread per output pixel, 64-wide acc.
// Writes f as fp32 NCHW (for bilinear gathers) and f16 NHWC (WMMA im2col).
// ---------------------------------------------------------------------------
__global__ __launch_bounds__(256) void conv_bb_bn_relu_k(
    const float* __restrict__ x, const float* __restrict__ w_bb,
    const float* __restrict__ b_bb,
    const float* __restrict__ bn_g, const float* __restrict__ bn_b,
    const float* __restrict__ bn_m, const float* __restrict__ bn_v,
    float* __restrict__ feat, _Float16* __restrict__ feath) {
  __shared__ float wl[147 * 64];           // [tap][oc], tap = c*49 + ky*7 + kx
  for (int i = threadIdx.x; i < 147 * 64; i += 256) {
    int tap = i >> 6, oc = i & 63;
    wl[i] = w_bb[oc * 147 + tap];
  }
  __syncthreads();

  const int pix = blockIdx.x * 256 + threadIdx.x;
  const int b = pix / SP, sp = pix % SP;
  const int oh = sp / Wc, ow = sp % Wc;
  float acc[64];
#pragma unroll
  for (int o = 0; o < 64; ++o) acc[o] = 0.f;

  int tap = 0;
  for (int c = 0; c < 3; ++c) {
    for (int ky = 0; ky < 7; ++ky) {
      const int ih = oh * 2 - 3 + ky;
      for (int kx = 0; kx < 7; ++kx, ++tap) {
        const int iw = ow * 2 - 3 + kx;
        float xv = 0.f;
        if (ih >= 0 && ih < 224 && iw >= 0 && iw < 224)
          xv = x[((b * 3 + c) * 224 + ih) * 224 + iw];
        const float* wp = &wl[tap * 64];
#pragma unroll 16
        for (int o = 0; o < 64; ++o) acc[o] = fmaf(xv, wp[o], acc[o]);
      }
    }
  }
#pragma unroll 4
  for (int o = 0; o < 64; ++o) {
    float s = bn_g[o] * rsqrtf(bn_v[o] + 1e-5f);
    float v = (acc[o] + b_bb[o] - bn_m[o]) * s + bn_b[o];
    v = fmaxf(v, 0.f);
    feat[((size_t)(b * 64 + o)) * SP + sp] = v;
    feath[(size_t)pix * 64 + o] = (_Float16)v;
  }
}

// ---------------------------------------------------------------------------
// K2: fused offset(18)+mask(9) 3x3 conv as a WMMA GEMM, M=32 (padded), K=576.
// Each wave owns 16 pixels; 2 accumulators cover M=0..31. B operand comes
// straight from f16 NHWC (channels contiguous per tap pixel).
// ---------------------------------------------------------------------------
__global__ __launch_bounds__(256) void conv_offmask_wmma_k(
    const _Float16* __restrict__ feath, const _Float16* __restrict__ Aom,
    const float* __restrict__ b_off, const float* __restrict__ b_msk,
    float* __restrict__ offs, float* __restrict__ maskb) {
  const int lane = threadIdx.x & 31;
  const int wave = threadIdx.x >> 5;
  const int col = lane & 15, half = lane >> 4;
  const int pix = blockIdx.x * 128 + wave * 16 + col;
  const int b = pix / SP, sp = pix % SP;
  const int oh = sp / Wc, ow = sp % Wc;

  v8f acc0 = {}, acc1 = {};
  for (int chunk = 0; chunk < NCHUNK; ++chunk) {
    const int tap = chunk >> 1;
    const int ky = tap / 3, kx = tap % 3;
    const int c0 = (chunk & 1) * 32 + half * 16;
    const int y = oh + ky - 1, xw = ow + kx - 1;
    v16h bm = {};
    if (y >= 0 && y < Hc && xw >= 0 && xw < Wc)
      bm = *(const v16h*)&feath[(((size_t)b * Hc + y) * Wc + xw) * 64 + c0];
    v16h a0 = *(const v16h*)&Aom[((size_t)(0 * NCHUNK + chunk) * 32 + lane) * 16];
    v16h a1 = *(const v16h*)&Aom[((size_t)(1 * NCHUNK + chunk) * 32 + lane) * 16];
    acc0 = __builtin_amdgcn_wmma_f32_16x16x32_f16(false, a0, false, bm, (short)0, acc0, false, false);
    acc1 = __builtin_amdgcn_wmma_f32_16x16x32_f16(false, a1, false, bm, (short)0, acc1, false, false);
  }
#pragma unroll
  for (int r = 0; r < 8; ++r) {
    const int M0 = half * 8 + r;                       // 0..15: offset rows
    offs[((size_t)b * 18 + M0) * SP + sp] = acc0[r] + b_off[M0];
    const int M1 = 16 + half * 8 + r;                  // 16..31
    if (M1 < 18) {
      offs[((size_t)b * 18 + M1) * SP + sp] = acc1[r] + b_off[M1];
    } else if (M1 < 27) {
      float v = acc1[r] + b_msk[M1 - 18];
      maskb[((size_t)b * 9 + (M1 - 18)) * SP + sp] = 1.f / (1.f + __expf(-v));
    }
  }
}

// ---------------------------------------------------------------------------
// K3: DCNv2 + pooling.
//  - Wave 0 kicks a single TDM tensor_load_to_lds of the whole packed w_dcn
//    (147456 B, 18432 x 8B elements, 1D tile) into LDS, overlapped with the
//    bilinear gather phase; s_wait_tensorcnt 0 before the block barrier.
//  - Phase 1 builds a [64 pixels x 576 K] sampled f16 tile in LDS.
//  - Phase 2: 8x4-tile WMMA GEMM, A from LDS (ds_load_b128), B from LDS;
//    outputs reduced over pixels (shfl_xor) -> LDS -> global atomics.
//  y (51 MB) is never written to memory.
// ---------------------------------------------------------------------------
__global__ __launch_bounds__(256) void dcn_pool_wmma_k(
    const float* __restrict__ feat, const float* __restrict__ offs,
    const float* __restrict__ maskb, const _Float16* __restrict__ Adcn,
    float* __restrict__ pooled) {
  extern __shared__ _Float16 smp[];        // [TILE_N][LSTR] then Aw[73728]
  __shared__ float psum[OC_DCN];
  _Float16* awlds = smp + TILE_N * LSTR;
  const int tid = threadIdx.x;
  const int wgBase = blockIdx.x * TILE_N;
  const int b = wgBase / SP;               // uniform: 64 divides 12544
  const int spBase = wgBase % SP;
  if (tid < OC_DCN) psum[tid] = 0.f;

  // ---- TDM: DMA packed weights into LDS (one op per workgroup, wave 0) ----
  if (tid < 32) {
    unsigned ldsOff = (unsigned)(unsigned long long)(void*)awlds;
    unsigned long long ga = (unsigned long long)(const void*)Adcn;
    v4u g0;
    g0[0] = 1u;                                        // count=1, is_restore=0
    g0[1] = ldsOff;                                    // lds_addr
    g0[2] = (unsigned)(ga & 0xFFFFFFFFu);              // global_addr[31:0]
    g0[3] = (unsigned)((ga >> 32) & 0x1FFFFFFu) | (2u << 30);  // addr[56:32]|type=2
    v8u g1;
    g1[0] = (3u << 16);                                // wg_mask=0, data_size=8B
    g1[1] = ((unsigned)AW_QWORDS & 0xFFFFu) << 16;     // tensor_dim0[15:0]
    g1[2] = ((unsigned)AW_QWORDS >> 16) | (1u << 16);  // tensor_dim0[31:16], tensor_dim1=1
    g1[3] = ((unsigned)AW_QWORDS & 0xFFFFu) << 16;     // tile_dim0 = 18432
    g1[4] = 1u;                                        // tile_dim1=1, tile_dim2=0
    g1[5] = (unsigned)AW_QWORDS;                       // tensor_dim0_stride[31:0]
    g1[6] = 0u;
    g1[7] = 0u;
    v4u g2 = {0u, 0u, 0u, 0u};
    v4u g3 = {0u, 0u, 0u, 0u};
    asm volatile("tensor_load_to_lds %0, %1, %2, %3"
                 :: "s"(g0), "s"(g1), "s"(g2), "s"(g3)
                 : "memory");
  }

  // ---- phase 1: bilinear sample into LDS (K order = tap*64 + c) ----
  for (int p = tid; p < 9 * TILE_N; p += 256) {
    const int tap = p >> 6, n = p & 63;
    const int sp = spBase + n;
    const int oh = sp / Wc, ow = sp % Wc;
    const int ky = tap / 3, kx = tap % 3;
    const float offy = offs[((size_t)b * 18 + 2 * tap + 0) * SP + sp];
    const float offx = offs[((size_t)b * 18 + 2 * tap + 1) * SP + sp];
    const float m    = maskb[((size_t)b * 9 + tap) * SP + sp];
    const float py = (float)(oh - 1 + ky) + offy;
    const float px = (float)(ow - 1 + kx) + offx;
    const float fy = floorf(py), fx = floorf(px);
    const int y0 = (int)fy, x0 = (int)fx, y1 = y0 + 1, x1 = x0 + 1;
    const float wy1 = py - fy, wx1 = px - fx;
    const float wy0 = 1.f - wy1, wx0 = 1.f - wx1;
    const float vy0 = (y0 >= 0 && y0 < Hc) ? 1.f : 0.f;
    const float vy1 = (y1 >= 0 && y1 < Hc) ? 1.f : 0.f;
    const float vx0 = (x0 >= 0 && x0 < Wc) ? 1.f : 0.f;
    const float vx1 = (x1 >= 0 && x1 < Wc) ? 1.f : 0.f;
    const int cy0 = min(max(y0, 0), Hc - 1), cy1 = min(max(y1, 0), Hc - 1);
    const int cx0 = min(max(x0, 0), Wc - 1), cx1 = min(max(x1, 0), Wc - 1);
    const float w00 = wy0 * wx0 * vy0 * vx0 * m, w01 = wy0 * wx1 * vy0 * vx1 * m;
    const float w10 = wy1 * wx0 * vy1 * vx0 * m, w11 = wy1 * wx1 * vy1 * vx1 * m;
    const float* fb = feat + (size_t)b * 64 * SP;
    const size_t i00 = (size_t)cy0 * Wc + cx0, i01 = (size_t)cy0 * Wc + cx1;
    const size_t i10 = (size_t)cy1 * Wc + cx0, i11 = (size_t)cy1 * Wc + cx1;
    _Float16* dst = &smp[n * LSTR + tap * 64];
    for (int c = 0; c < 64; ++c) {
      const float* fc = fb + (size_t)c * SP;
      if ((c & 15) == 0) __builtin_prefetch(fc + 4 * SP + i00, 0, 1);
      float v = w00 * fc[i00] + w01 * fc[i01] + w10 * fc[i10] + w11 * fc[i11];
      dst[c] = (_Float16)v;
    }
  }
  if (tid < 32) __builtin_amdgcn_s_wait_tensorcnt(0);
  __syncthreads();

  // ---- phase 2: WMMA GEMM (M=128, N=64, K=576) + pixel reduction ----
  const int lane = tid & 31, wave = tid >> 5;
  const int ntile = wave & 3, mg = wave >> 2;     // mg 0/1 -> mtiles mg*4..mg*4+3
  const int col = lane & 15, half = lane >> 4;
  v8f acc[4] = {};
  for (int chunk = 0; chunk < NCHUNK; ++chunk) {
    const int n = ntile * 16 + col;
    const int k0 = chunk * 32 + half * 16;
    v16h bm = *(const v16h*)&smp[n * LSTR + k0];
#pragma unroll
    for (int t = 0; t < 4; ++t) {
      v16h a = *(const v16h*)&awlds[((size_t)((mg * 4 + t) * NCHUNK + chunk) * 32 + lane) * 16];
      acc[t] = __builtin_amdgcn_wmma_f32_16x16x32_f16(false, a, false, bm, (short)0, acc[t], false, false);
    }
  }
#pragma unroll
  for (int t = 0; t < 4; ++t) {
#pragma unroll
    for (int r = 0; r < 8; ++r) {
      float v = acc[t][r];
      v += __shfl_xor(v, 1);
      v += __shfl_xor(v, 2);
      v += __shfl_xor(v, 4);
      v += __shfl_xor(v, 8);
      if (col == 0) {
        const int M = mg * 64 + t * 16 + half * 8 + r;
        atomicAdd(&psum[M], v);
      }
    }
  }
  __syncthreads();
  if (tid < OC_DCN) atomicAdd(&pooled[b * OC_DCN + tid], psum[tid]);
}

// ---------------------------------------------------------------------------
// K4: mean (+ b_dcn applied once, post-mean) and the 128->1000 FC. Tiny.
// ---------------------------------------------------------------------------
__global__ void fc_k(const float* __restrict__ pooled, const float* __restrict__ b_dcn,
                     const float* __restrict__ w_fc, const float* __restrict__ b_fc,
                     float* __restrict__ out) {
  int i = blockIdx.x * blockDim.x + threadIdx.x;
  if (i >= B_ * 1000) return;
  const int b = i / 1000, j = i % 1000;
  float acc = b_fc[j];
  const float* wr = w_fc + (size_t)j * 128;
  const float* pr = pooled + b * 128;
  const float inv = 1.f / (float)SP;
#pragma unroll 8
  for (int m = 0; m < 128; ++m)
    acc += (pr[m] * inv + b_dcn[m]) * wr[m];
  out[i] = acc;
}

extern "C" void kernel_launch(void* const* d_in, const int* in_sizes, int n_in,
                              void* d_out, int out_size, void* d_ws, size_t ws_size,
                              hipStream_t stream) {
  (void)in_sizes; (void)n_in; (void)out_size; (void)ws_size;
  const float* x     = (const float*)d_in[0];
  const float* w_bb  = (const float*)d_in[1];
  const float* b_bb  = (const float*)d_in[2];
  const float* bn_g  = (const float*)d_in[3];
  const float* bn_b  = (const float*)d_in[4];
  const float* bn_m  = (const float*)d_in[5];
  const float* bn_v  = (const float*)d_in[6];
  const float* w_off = (const float*)d_in[7];
  const float* b_off = (const float*)d_in[8];
  const float* w_msk = (const float*)d_in[9];
  const float* b_msk = (const float*)d_in[10];
  const float* w_dcn = (const float*)d_in[11];
  const float* b_dcn = (const float*)d_in[12];
  const float* w_fc  = (const float*)d_in[13];
  const float* b_fc  = (const float*)d_in[14];

  char* ws = (char*)d_ws;
  size_t off = 0;
  auto carve = [&](size_t bytes) {
    char* p = ws + off;
    off += (bytes + 255) & ~(size_t)255;
    return (void*)p;
  };
  float*    feat   = (float*)carve(sizeof(float) * (size_t)NPIX * 64);       // 25.7 MB
  _Float16* feath  = (_Float16*)carve(sizeof(_Float16) * (size_t)NPIX * 64); // 12.8 MB
  _Float16* Aom    = (_Float16*)carve(sizeof(_Float16) * 2 * NCHUNK * 512);
  _Float16* Adcn   = (_Float16*)carve(sizeof(_Float16) * AW_HALVES);
  float*    offs   = (float*)carve(sizeof(float) * (size_t)B_ * 18 * SP);    // 7.2 MB
  float*    maskb  = (float*)carve(sizeof(float) * (size_t)B_ * 9 * SP);     // 3.6 MB
  float*    pooled = (float*)carve(sizeof(float) * B_ * OC_DCN);

  pack_weights_k<<<(8 * NCHUNK * 512 + 255) / 256, 256, 0, stream>>>(
      w_off, w_msk, w_dcn, Aom, Adcn, pooled);
  conv_bb_bn_relu_k<<<NPIX / 256, 256, 0, stream>>>(
      x, w_bb, b_bb, bn_g, bn_b, bn_m, bn_v, feat, feath);
  conv_offmask_wmma_k<<<NPIX / 128, 256, 0, stream>>>(
      feath, Aom, b_off, b_msk, offs, maskb);
  const size_t lds_bytes = (size_t)(TILE_N * LSTR + AW_HALVES) * sizeof(_Float16); // 223232
  dcn_pool_wmma_k<<<NPIX / TILE_N, 256, lds_bytes, stream>>>(
      feat, offs, maskb, Adcn, pooled);
  fc_k<<<(B_ * 1000 + 255) / 256, 256, 0, stream>>>(
      pooled, b_dcn, w_fc, b_fc, (float*)d_out);
}
